// BatchedVQLAMDecoder_5145370821030
// MI455X (gfx1250) — matrix operand order repaired and checked by reading, not compiled, not run on hardware
//
#include <hip/hip_runtime.h>
#include <hip/hip_bf16.h>
#include <math.h>

// Problem constants (B=1)
#define T_      8
#define N_      256
#define D_      1024
#define CD_     128
#define E_      512
#define H_      8
#define DH_     64
#define DEPTH_  3
#define CHUNK_  515          // 2*N_+3
#define L_      4120         // T_*CHUNK_
#define E3_     1536
#define E4_     2048

#define USE_ASYNC_LDS 1      // gfx1250 global_load_async_to_lds_b128 staging

typedef __bf16 bf16_t;
typedef bf16_t v16bf __attribute__((ext_vector_type(16)));
typedef float  v8f   __attribute__((ext_vector_type(8)));

__device__ __forceinline__ bf16_t f2bf(float f) {
    union { float f; unsigned u; } a; a.f = f;
    unsigned r = a.u + 0x7FFFu + ((a.u >> 16) & 1u);   // round-to-nearest-even
    unsigned short h = (unsigned short)(r >> 16);
    return __builtin_bit_cast(bf16_t, h);
}

#define WMMA_BF16(a, b, c) \
    __builtin_amdgcn_wmma_f32_16x16x32_bf16(false, (a), false, (b), (short)0, (c), false, false)

// ---- WMMA fragment loaders (contiguous-k layouts, wave32) -------------------
// A 16x32 bf16: lane m=lane&15, kbase=(lane>>4)*16; elems [0..15] = K kb..kb+15
// (contiguous, matching the B fragments)
__device__ __forceinline__ v16bf load_frag_A16x32(const bf16_t* s, int mbase, int stride, int lane) {
    int mm = mbase + (lane & 15);
    int kb = (lane >> 4) * 16;
    const bf16_t* p0 = s + mm * stride + kb;
    const bf16_t* p1 = p0 + 8;
    v16bf a;
#pragma unroll
    for (int j = 0; j < 8; ++j) { a[j] = p0[j]; a[8 + j] = p1[j]; }
    return a;
}

// B 32x16 from LDS tile stored TRANSPOSED [n][k] (row stride = `stride` Ks):
// lane n=lane&15, kbase=(lane>>4)*16, elems j -> K=kbase+j : fully contiguous
__device__ __forceinline__ v16bf load_frag_Bn(const bf16_t* s, int nbase, int stride, int lane) {
    const bf16_t* p = s + (nbase + (lane & 15)) * stride + (lane >> 4) * 16;
    v16bf b;
#pragma unroll
    for (int j = 0; j < 16; ++j) b[j] = p[j];
    return b;
}

// B for Q@K^T: contraction over dims, columns = keys; K tile LDS is [key][d]
// (stride 64) so per-lane dims are contiguous already.
__device__ __forceinline__ v16bf load_frag_Bt(const bf16_t* sK, int keyoff, int doff, int lane) {
    const bf16_t* p = sK + (keyoff + (lane & 15)) * 64 + doff + (lane >> 4) * 16;
    v16bf b;
#pragma unroll
    for (int j = 0; j < 16; ++j) b[j] = p[j];
    return b;
}

// ---- generic bf16 WMMA GEMM: out = epi(A[MxK] @ Bw[KxN] + bias) -----------
// block: 256 thr / 8 waves, tile 128x64, K-step 32; wave = 32x32 (2x2 WMMA)
template <bool GELU, bool RES, bool BF16OUT>
__global__ __launch_bounds__(256) void gemm_bf16_kernel(
    const bf16_t* __restrict__ A, const bf16_t* __restrict__ Bw,
    const float* __restrict__ bias, const float* resid,
    float* outF, bf16_t* outB, int M, int Nn, int K)
{
    __shared__ bf16_t sA[128 * 32];      // [m][k]
    __shared__ bf16_t sBt[64 * 32];      // [n][k] (transposed for contiguous frags)
    const int tid    = threadIdx.x;
    const int lane   = tid & 31;
    const int wave   = tid >> 5;
    const int waveM  = (wave >> 1) * 32;
    const int waveN  = (wave & 1) * 32;
    const int blockM = blockIdx.x * 128;
    const int blockN = blockIdx.y * 64;
    const bf16_t bz  = f2bf(0.0f);
    const bool fullM = (blockM + 128 <= M);
#if USE_ASYNC_LDS
    const unsigned ldsA0 = (unsigned)(size_t)&sA[0];
#endif

    v8f acc[2][2];
    acc[0][0] = {}; acc[0][1] = {}; acc[1][0] = {}; acc[1][1] = {};

    for (int kt = 0; kt < K; kt += 32) {
        if (fullM) {
#if USE_ASYNC_LDS
            // async global->LDS staging of the 128x32 A tile (ASYNCcnt path):
            // 512 chunks of 16B; 4 chunks per 64B row; lds byte = chunk*16
#pragma unroll
            for (int i = 0; i < 2; ++i) {
                int c = i * 256 + tid;
                int row = c >> 2;
                unsigned lds  = ldsA0 + (unsigned)c * 16u;
                unsigned goff = (unsigned)((((size_t)(blockM + row)) * K + kt + (size_t)(c & 3) * 8) * 2);
                asm volatile("global_load_async_to_lds_b128 %0, %1, %2"
                             :: "v"(lds), "v"(goff), "s"(A) : "memory");
            }
#else
#pragma unroll
            for (int i = 0; i < 16; ++i) {
                int e = i * 256 + tid;
                int r = e >> 5, c = e & 31;
                sA[e] = A[(size_t)(blockM + r) * K + kt + c];
            }
#endif
        } else {
#pragma unroll
            for (int i = 0; i < 16; ++i) {       // guarded tail path
                int e = i * 256 + tid;
                int r = e >> 5, c = e & 31;
                int gr = blockM + r;
                sA[e] = (gr < M) ? A[(size_t)gr * K + kt + c] : bz;
            }
        }
#pragma unroll
        for (int i = 0; i < 8; ++i) {            // B tile 32x64 -> sBt[n][k]
            int e = i * 256 + tid;
            int r = e >> 6, c = e & 63;
            sBt[c * 32 + r] = Bw[(size_t)(kt + r) * Nn + blockN + c];
        }
        if (kt + 32 < K)                         // CDNA5 global_prefetch_b8
            __builtin_prefetch(&Bw[(size_t)(kt + 32 + (tid >> 3)) * Nn + blockN + (tid & 7) * 8], 0, 1);
#if USE_ASYNC_LDS
        asm volatile("s_wait_asynccnt 0x0" ::: "memory");
#endif
        __syncthreads();

        v16bf a0 = load_frag_A16x32(sA, waveM, 32, lane);
        v16bf a1 = load_frag_A16x32(sA, waveM + 16, 32, lane);
        v16bf b0 = load_frag_Bn(sBt, waveN, 32, lane);
        v16bf b1 = load_frag_Bn(sBt, waveN + 16, 32, lane);
        acc[0][0] = WMMA_BF16(a0, b0, acc[0][0]);
        acc[0][1] = WMMA_BF16(a0, b1, acc[0][1]);
        acc[1][0] = WMMA_BF16(a1, b0, acc[1][0]);
        acc[1][1] = WMMA_BF16(a1, b1, acc[1][1]);
        __syncthreads();
    }

    const int nn = lane & 15;
    const int hf = lane >> 4;
#pragma unroll
    for (int ti = 0; ti < 2; ++ti)
#pragma unroll
        for (int tj = 0; tj < 2; ++tj)
#pragma unroll
            for (int r = 0; r < 8; ++r) {
                int row = blockM + waveM + ti * 16 + r + 8 * hf;
                int col = blockN + waveN + tj * 16 + nn;
                if (row < M) {
                    float v = acc[ti][tj][r] + bias[col];
                    if (GELU) {
                        float x3 = v * v * v;
                        v = 0.5f * v * (1.0f + tanhf(0.7978845608028654f * (v + 0.044715f * x3)));
                    }
                    if (RES) v += resid[(size_t)row * Nn + col];
                    if (BF16OUT) outB[(size_t)row * Nn + col] = f2bf(v);
                    else         outF[(size_t)row * Nn + col] = v;
                }
            }
}

// ---- flash attention: 4 waves/block share K/V tiles; block-causal ----------
__device__ __forceinline__ float half16_max(float v) {
#pragma unroll
    for (int m = 1; m < 16; m <<= 1) v = fmaxf(v, __shfl_xor(v, m, 32));
    return v;
}
__device__ __forceinline__ float half16_sum(float v) {
#pragma unroll
    for (int m = 1; m < 16; m <<= 1) v += __shfl_xor(v, m, 32);
    return v;
}

__global__ __launch_bounds__(128) void attention_kernel(const bf16_t* __restrict__ qkv,
                                                        bf16_t* __restrict__ obuf)
{
    __shared__ bf16_t sQ[64 * 64];       // 4 waves x 16 rows, [m][d]
    __shared__ bf16_t sK[32 * 64];       // [key][d]
    __shared__ bf16_t sVt[64 * 32];      // [d][key]  (transposed for frags)
    __shared__ bf16_t sP[4][16 * 32];    // per-wave P staging
    const int tid    = threadIdx.x;
    const int lane   = tid & 31;
    const int wv     = tid >> 5;
    const int head   = blockIdx.y;
    const int blkrow = blockIdx.x * 64;
    const int qbase  = blkrow + wv * 16;
    const bf16_t bz  = f2bf(0.0f);

#pragma unroll
    for (int i = 0; i < 32; ++i) {               // Q tile 64x64 cooperative
        int e = i * 128 + tid;
        int r = e >> 6, d = e & 63;
        int gq = blkrow + r;
        sQ[e] = (gq < L_) ? qkv[(size_t)gq * E3_ + head * 64 + d] : bz;
    }
    __syncthreads();
    v16bf aq0 = load_frag_A16x32(sQ + wv * (16 * 64), 0, 64, lane);       // dims 0..31
    v16bf aq1 = load_frag_A16x32(sQ + wv * (16 * 64) + 32, 0, 64, lane);  // dims 32..63

    v8f O0 = {}, O1 = {}, O2 = {}, O3 = {};
    float mrow[8], lrow[8];
#pragma unroll
    for (int r = 0; r < 8; ++r) { mrow[r] = -1e30f; lrow[r] = 0.0f; }

    const int nn = lane & 15, hf = lane >> 4;
    int lastrow = blkrow + 63; if (lastrow >= L_) lastrow = L_ - 1;
    const int kend = (lastrow / CHUNK_ + 1) * CHUNK_;   // block-uniform

    for (int kt = 0; kt < kend; kt += 32) {
        __syncthreads();
#pragma unroll
        for (int i = 0; i < 16; ++i) {           // K,V tiles cooperative
            int e = i * 128 + tid;
            int r = e >> 6, d = e & 63;
            int gk = kt + r;
            bool ok = gk < L_;
            sK[e]           = ok ? qkv[(size_t)gk * E3_ + E_ + head * 64 + d]     : bz;
            sVt[d * 32 + r] = ok ? qkv[(size_t)gk * E3_ + 2 * E_ + head * 64 + d] : bz;
        }
        __syncthreads();

        v8f s0 = {}, s1 = {};
        {
            v16bf b00 = load_frag_Bt(sK, 0, 0, lane);
            v16bf b01 = load_frag_Bt(sK, 0, 32, lane);
            v16bf b10 = load_frag_Bt(sK, 16, 0, lane);
            v16bf b11 = load_frag_Bt(sK, 16, 32, lane);
            s0 = WMMA_BF16(aq0, b00, s0); s0 = WMMA_BF16(aq1, b01, s0);
            s1 = WMMA_BF16(aq0, b10, s1); s1 = WMMA_BF16(aq1, b11, s1);
        }

#pragma unroll
        for (int r = 0; r < 8; ++r) {            // online softmax per row
            int m = r + 8 * hf;
            int q = qbase + m;
            int kmaxr = (q < L_) ? ((q / CHUNK_) + 1) * CHUNK_ : 1;
            float v0 = s0[r] * 0.125f; if (kt + nn >= kmaxr)      v0 = -1e9f;
            float v1 = s1[r] * 0.125f; if (kt + 16 + nn >= kmaxr) v1 = -1e9f;
            float tmax  = half16_max(fmaxf(v0, v1));
            float mnew  = fmaxf(mrow[r], tmax);
            float alpha = __expf(mrow[r] - mnew);
            float p0 = __expf(v0 - mnew), p1 = __expf(v1 - mnew);
            lrow[r] = lrow[r] * alpha + half16_sum(p0 + p1);
            mrow[r] = mnew;
            O0[r] *= alpha; O1[r] *= alpha; O2[r] *= alpha; O3[r] *= alpha;
            sP[wv][m * 32 + nn]      = f2bf(p0);
            sP[wv][m * 32 + 16 + nn] = f2bf(p1);
        }
        __syncthreads();
        v16bf aP  = load_frag_A16x32(sP[wv], 0, 32, lane);
        v16bf bv0 = load_frag_Bn(sVt, 0,  32, lane);
        v16bf bv1 = load_frag_Bn(sVt, 16, 32, lane);
        v16bf bv2 = load_frag_Bn(sVt, 32, 32, lane);
        v16bf bv3 = load_frag_Bn(sVt, 48, 32, lane);
        O0 = WMMA_BF16(aP, bv0, O0); O1 = WMMA_BF16(aP, bv1, O1);
        O2 = WMMA_BF16(aP, bv2, O2); O3 = WMMA_BF16(aP, bv3, O3);
    }

#pragma unroll
    for (int r = 0; r < 8; ++r) {
        int m = r + 8 * hf;
        int row = qbase + m;
        if (row < L_) {
            float inv = 1.0f / lrow[r];
            size_t base = (size_t)row * E_ + head * 64;
            obuf[base + 0  + nn] = f2bf(O0[r] * inv);
            obuf[base + 16 + nn] = f2bf(O1[r] * inv);
            obuf[base + 32 + nn] = f2bf(O2[r] * inv);
            obuf[base + 48 + nn] = f2bf(O3[r] * inv);
        }
    }
}

// ---- LayerNorm (affine), one block per row, f32 in -> bf16 out -------------
__global__ __launch_bounds__(256) void ln_kernel(const float* __restrict__ x,
                                                 const float* __restrict__ g,
                                                 const float* __restrict__ b,
                                                 bf16_t* __restrict__ y)
{
    const int row = blockIdx.x, tid = threadIdx.x;
    const float* xr = x + (size_t)row * E_;
    float v0 = xr[tid], v1 = xr[tid + 256];
    __shared__ float red[256];
    red[tid] = v0 + v1; __syncthreads();
    for (int s = 128; s > 0; s >>= 1) { if (tid < s) red[tid] += red[tid + s]; __syncthreads(); }
    float mean = red[0] * (1.0f / E_); __syncthreads();
    float d0 = v0 - mean, d1 = v1 - mean;
    red[tid] = d0 * d0 + d1 * d1; __syncthreads();
    for (int s = 128; s > 0; s >>= 1) { if (tid < s) red[tid] += red[tid + s]; __syncthreads(); }
    float rstd = rsqrtf(red[0] * (1.0f / E_) + 1e-5f);
    y[(size_t)row * E_ + tid]       = f2bf(d0 * rstd * g[tid] + b[tid]);
    y[(size_t)row * E_ + tid + 256] = f2bf(d1 * rstd * g[tid + 256] + b[tid + 256]);
}

// ---- seq assembly: LN(z) | LN(c) | q, + pos_embed, f32 ---------------------
__global__ __launch_bounds__(256) void assemble_kernel(const float* __restrict__ zraw,
                                                       const float* __restrict__ craw,
                                                       const float* __restrict__ qemb,
                                                       const float* __restrict__ pos,
                                                       float* __restrict__ seq)
{
    const int l = blockIdx.x, tid = threadIdx.x;
    const int t = l / CHUNK_, j = l % CHUNK_;
    const float* src; bool doLN = true;
    if (j < N_)          src = zraw + (size_t)(t * N_ + j) * E_;
    else if (j < N_ + 3) src = craw + (size_t)(t * 3 + (j - N_)) * E_;
    else { src = qemb + (size_t)(j - N_ - 3) * E_; doLN = false; }
    float v0 = src[tid], v1 = src[tid + 256];
    float o0 = v0, o1 = v1;
    __shared__ float red[256];
    if (doLN) {
        red[tid] = v0 + v1; __syncthreads();
        for (int s = 128; s > 0; s >>= 1) { if (tid < s) red[tid] += red[tid + s]; __syncthreads(); }
        float mean = red[0] * (1.0f / E_); __syncthreads();
        float d0 = v0 - mean, d1 = v1 - mean;
        red[tid] = d0 * d0 + d1 * d1; __syncthreads();
        for (int s = 128; s > 0; s >>= 1) { if (tid < s) red[tid] += red[tid + s]; __syncthreads(); }
        float rstd = rsqrtf(red[0] * (1.0f / E_) + 1e-5f);
        o0 = d0 * rstd; o1 = d1 * rstd;
    }
    seq[(size_t)l * E_ + tid]       = o0 + pos[(size_t)l * E_ + tid];
    seq[(size_t)l * E_ + tid + 256] = o1 + pos[(size_t)l * E_ + tid + 256];
}

__global__ void gather_pred_kernel(const bf16_t* __restrict__ seqln, bf16_t* __restrict__ pred)
{
    int idx = blockIdx.x * 256 + threadIdx.x;
    if (idx >= T_ * N_ * E_) return;
    int m = idx >> 9, e = idx & 511;
    int t = m >> 8, n = m & 255;
    int srow = t * CHUNK_ + N_ + 3 + n;
    pred[idx] = seqln[(size_t)srow * E_ + e];
}

__global__ void cvt_kernel(const float* __restrict__ x, bf16_t* __restrict__ y, int n)
{
    int i = blockIdx.x * 256 + threadIdx.x;
    if (i < n) y[i] = f2bf(x[i]);
}

// ---------------------------------------------------------------------------
extern "C" void kernel_launch(void* const* d_in, const int* in_sizes, int n_in,
                              void* d_out, int out_size, void* d_ws, size_t ws_size,
                              hipStream_t stream)
{
    (void)in_sizes; (void)n_in; (void)out_size; (void)ws_size;
    const float* z_past   = (const float*)d_in[0];
    const float* code_emb = (const float*)d_in[1];
    const float* patch_w  = (const float*)d_in[2];
    const float* patch_b  = (const float*)d_in[3];
    const float* code_w   = (const float*)d_in[4];
    const float* code_b   = (const float*)d_in[5];
    const float* query_e  = (const float*)d_in[6];
    const float* pos_e    = (const float*)d_in[7];
    const float* ln1_g    = (const float*)d_in[8];
    const float* ln1_b    = (const float*)d_in[9];
    const float* qkv_w    = (const float*)d_in[10];
    const float* qkv_b    = (const float*)d_in[11];
    const float* out_w    = (const float*)d_in[12];
    const float* out_b    = (const float*)d_in[13];
    const float* ln2_g    = (const float*)d_in[14];
    const float* ln2_b    = (const float*)d_in[15];
    const float* mlp_w1   = (const float*)d_in[16];
    const float* mlp_b1   = (const float*)d_in[17];
    const float* mlp_w2   = (const float*)d_in[18];
    const float* mlp_b2   = (const float*)d_in[19];
    const float* norm_g   = (const float*)d_in[20];
    const float* norm_b   = (const float*)d_in[21];
    const float* oproj_w  = (const float*)d_in[22];
    const float* oproj_b  = (const float*)d_in[23];

    char* p = (char*)d_ws;
    auto take = [&](size_t bytes) { char* q = p; p += (bytes + 255) & ~(size_t)255; return q; };
    float*  seq    = (float*) take((size_t)L_ * E_ * 4);
    bf16_t* hbuf   = (bf16_t*)take((size_t)L_ * E_ * 2);
    bf16_t* qkvb   = (bf16_t*)take((size_t)L_ * E3_ * 2);
    bf16_t* obuf   = (bf16_t*)take((size_t)L_ * E_ * 2);
    bf16_t* mbuf   = (bf16_t*)take((size_t)L_ * E4_ * 2);
    float*  zraw   = (float*) take((size_t)T_ * N_ * E_ * 4);
    float*  craw   = (float*) take((size_t)T_ * 3 * E_ * 4);
    bf16_t* predb  = (bf16_t*)take((size_t)T_ * N_ * E_ * 2);
    bf16_t* zp_bf  = (bf16_t*)take((size_t)T_ * N_ * D_ * 2);
    bf16_t* ce_bf  = (bf16_t*)take((size_t)T_ * 3 * CD_ * 2);
    bf16_t* patchW = (bf16_t*)take((size_t)D_ * E_ * 2);
    bf16_t* codeW  = (bf16_t*)take((size_t)CD_ * E_ * 2);
    bf16_t* qkvW   = (bf16_t*)take((size_t)DEPTH_ * E_ * E3_ * 2);
    bf16_t* outW   = (bf16_t*)take((size_t)DEPTH_ * E_ * E_ * 2);
    bf16_t* mlpW1  = (bf16_t*)take((size_t)DEPTH_ * E_ * E4_ * 2);
    bf16_t* mlpW2  = (bf16_t*)take((size_t)DEPTH_ * E4_ * E_ * 2);
    bf16_t* oprojW = (bf16_t*)take((size_t)E_ * D_ * 2);

    auto cvt = [&](const float* s, bf16_t* d, int n) {
        cvt_kernel<<<(n + 255) / 256, 256, 0, stream>>>(s, d, n);
    };
    cvt(z_past,   zp_bf,  T_ * N_ * D_);
    cvt(code_emb, ce_bf,  T_ * 3 * CD_);
    cvt(patch_w,  patchW, D_ * E_);
    cvt(code_w,   codeW,  CD_ * E_);
    cvt(qkv_w,    qkvW,   DEPTH_ * E_ * E3_);
    cvt(out_w,    outW,   DEPTH_ * E_ * E_);
    cvt(mlp_w1,   mlpW1,  DEPTH_ * E_ * E4_);
    cvt(mlp_w2,   mlpW2,  DEPTH_ * E4_ * E_);
    cvt(oproj_w,  oprojW, E_ * D_);

    auto gg = [](int M, int Nn) { return dim3((unsigned)((M + 127) / 128), (unsigned)((Nn + 63) / 64)); };

    // patch / code embed GEMMs -> f32 raw, then assemble seq (+LN, +pos)
    gemm_bf16_kernel<false, false, false><<<gg(T_ * N_, E_), 256, 0, stream>>>(
        zp_bf, patchW, patch_b, nullptr, zraw, nullptr, T_ * N_, E_, D_);
    gemm_bf16_kernel<false, false, false><<<gg(T_ * 3, E_), 256, 0, stream>>>(
        ce_bf, codeW, code_b, nullptr, craw, nullptr, T_ * 3, E_, CD_);
    assemble_kernel<<<L_, 256, 0, stream>>>(zraw, craw, query_e, pos_e, seq);

    for (int i = 0; i < DEPTH_; ++i) {
        ln_kernel<<<L_, 256, 0, stream>>>(seq, ln1_g + i * E_, ln1_b + i * E_, hbuf);
        gemm_bf16_kernel<false, false, true><<<gg(L_, E3_), 256, 0, stream>>>(
            hbuf, qkvW + (size_t)i * E_ * E3_, qkv_b + i * E3_, nullptr, nullptr, qkvb, L_, E3_, E_);
        attention_kernel<<<dim3((L_ + 63) / 64, H_), 128, 0, stream>>>(qkvb, obuf);
        gemm_bf16_kernel<false, true, false><<<gg(L_, E_), 256, 0, stream>>>(
            obuf, outW + (size_t)i * E_ * E_, out_b + i * E_, seq, seq, nullptr, L_, E_, E_);
        ln_kernel<<<L_, 256, 0, stream>>>(seq, ln2_g + i * E_, ln2_b + i * E_, hbuf);
        gemm_bf16_kernel<true, false, true><<<gg(L_, E4_), 256, 0, stream>>>(
            hbuf, mlpW1 + (size_t)i * E_ * E4_, mlp_b1 + i * E4_, nullptr, nullptr, mbuf, L_, E4_, E_);
        gemm_bf16_kernel<false, true, false><<<gg(L_, E_), 256, 0, stream>>>(
            mbuf, mlpW2 + (size_t)i * E4_ * E_, mlp_b2 + i * E_, seq, seq, nullptr, L_, E_, E4_);
    }

    ln_kernel<<<L_, 256, 0, stream>>>(seq, norm_g, norm_b, hbuf);
    gather_pred_kernel<<<(T_ * N_ * E_ + 255) / 256, 256, 0, stream>>>(hbuf, predb);
    gemm_bf16_kernel<false, false, false><<<gg(T_ * N_, D_), 256, 0, stream>>>(
        predb, oprojW, oproj_b, nullptr, (float*)d_out, nullptr, T_ * N_, D_, E_);
}
